// EmbeddingLayer_35192962024090
// MI455X (gfx1250) — compile-verified, matrix operand order checked by reading, not run on hardware
//
#include <hip/hip_runtime.h>
#include <hip/hip_bf16.h>

// ---------------------------------------------------------------------------
// EmbeddingLayer: per-feature MLP 1->100->64 + masked embedding replacement.
// B=16384, F=256, H=100 (pad K to 128), D=64.
// Hidden layer in packed bf16 VALU (v_pk_fma_bf16 + v_pk_max_num_bf16 via asm),
// GEMM via v_wmma_f32_16x16x32_bf16 with fp32 accumulation.
// ---------------------------------------------------------------------------

typedef __attribute__((ext_vector_type(16))) __bf16 v16bf;
typedef __attribute__((ext_vector_type(2)))  __bf16 v2bf;
typedef __attribute__((ext_vector_type(8)))  float  v8f;

union Frag {
    unsigned int u[8];
    v16bf        v;
};

union Pk2 {
    unsigned int u;
    v2bf         v;
};

#define FDIM 256
#define HDIM 100
#define DDIM 64
#define W2_STRIDE 80          // dword stride for packed W2 rows (bank-conflict free)
#define ROWS_PER_WG 2048      // B / gridDim.y
#define WG_THREADS 256        // 8 waves

__device__ __forceinline__ unsigned int f2bf_rne(float f) {
    unsigned int u = __float_as_uint(f);
    u += 0x7FFFu + ((u >> 16) & 1u);   // round-to-nearest-even to bf16
    return u >> 16;
}

// relu on a packed bf16 pair: single VOP3P v_pk_max_num_bf16 against inline 0
__device__ __forceinline__ unsigned int pk_relu_bf16(unsigned int h) {
    unsigned int r;
    asm("v_pk_max_num_bf16 %0, %1, 0" : "=v"(r) : "v"(h));
    return r;
}

__global__ __launch_bounds__(WG_THREADS)
void emb_mlp_wmma_kernel(const float* __restrict__ x,
                         const int*   __restrict__ x_mask,
                         const float* __restrict__ W1,
                         const float* __restrict__ b1,
                         const float* __restrict__ W2,
                         const float* __restrict__ b2,
                         const float* __restrict__ emb,
                         float*       __restrict__ out)
{
    __shared__ unsigned int sW2p[64 * W2_STRIDE]; // [k2][d] packed bf16 pairs (k, k+1)
    __shared__ unsigned int sW1p[64];             // packed bf16 {W1[2k2], W1[2k2+1]}
    __shared__ unsigned int sB1p[64];             // packed bf16 {b1[2k2], b1[2k2+1]}
    __shared__ float        sB2[DDIM];
    __shared__ float        sEmb[DDIM];
    __shared__ int          sMask[128];           // per-iteration row masks (8 waves x 16 rows)

    const int f    = blockIdx.x;
    const int tid  = threadIdx.x;
    const int lane = tid & 31;
    const int wave = tid >> 5;
    const int grp  = lane >> 4;   // half-wave group
    const int lr   = lane & 15;

    // ---- Stage W2[f] (100x64 fp32 -> 64x64 packed bf16-pair dwords, K zero-padded) ----
    const float* W2f = W2 + (size_t)f * HDIM * DDIM;
    for (int i = tid; i < 64 * 64; i += WG_THREADS) {
        const int k2 = i >> 6;
        const int d  = i & 63;
        const int k0 = 2 * k2;
        const float w0 = (k0     < HDIM) ? W2f[(size_t)k0       * DDIM + d] : 0.0f;
        const float w1 = (k0 + 1 < HDIM) ? W2f[(size_t)(k0 + 1) * DDIM + d] : 0.0f;
        sW2p[k2 * W2_STRIDE + d] = f2bf_rne(w0) | (f2bf_rne(w1) << 16);
    }
    // ---- Stage W1/b1 as packed bf16 pairs (zero-pad K>=100 -> relu(0)=0), b2, emb[2f] ----
    if (tid < 64) {
        const int k0 = 2 * tid;
        const float w0 = (k0     < HDIM) ? W1[(size_t)f * HDIM + k0    ] : 0.0f;
        const float w1 = (k0 + 1 < HDIM) ? W1[(size_t)f * HDIM + k0 + 1] : 0.0f;
        const float c0 = (k0     < HDIM) ? b1[(size_t)f * HDIM + k0    ] : 0.0f;
        const float c1 = (k0 + 1 < HDIM) ? b1[(size_t)f * HDIM + k0 + 1] : 0.0f;
        sW1p[tid] = f2bf_rne(w0) | (f2bf_rne(w1) << 16);
        sB1p[tid] = f2bf_rne(c0) | (f2bf_rne(c1) << 16);
        sB2[tid]  = b2[(size_t)f * DDIM + tid];
        sEmb[tid] = emb[(size_t)(2 * f) * DDIM + tid];
    }
    __syncthreads();

    // ---- Loop-invariant fragments/tables into registers ----
    // ISA 16-bit fragment K mapping: kbase = 2*(v&3) + 16*(v>>2) + 8*grp  (k2 = kbase/2)
    Frag bfr[16]; // B fragments (W2), [s*4 + n]
    Pk2  w1r[32]; // packed W1 pairs, [s*8 + v]
    Pk2  b1r[32]; // packed b1 pairs, [s*8 + v]
#pragma unroll
    for (int s = 0; s < 4; ++s) {
#pragma unroll
        for (int v = 0; v < 8; ++v) {
            const int k2 = (v & 3) + ((v >> 2) << 3) + (grp << 2) + (s << 4);
            w1r[s * 8 + v].u = sW1p[k2];
            b1r[s * 8 + v].u = sB1p[k2];
#pragma unroll
            for (int n = 0; n < 4; ++n)
                bfr[s * 4 + n].u[v] = sW2p[k2 * W2_STRIDE + n * 16 + lr];
        }
    }

    const size_t chunkBase = (size_t)blockIdx.y * ROWS_PER_WG;

    for (int it = 0; it < ROWS_PER_WG / 128; ++it) {
        const size_t base = chunkBase + (size_t)it * 128 + (size_t)wave * 16;
        const size_t row  = base + lr;

        // x column gather (L2-resident; lanes L and L+16 share an address)
        const float xv = x[row * FDIM + f];
        const int   mv = x_mask[row * FDIM + f];
        if (lane < 16) sMask[wave * 16 + lane] = mv; // same-wave DS ordering suffices

        // splat x to a packed bf16 pair
        Pk2 xs;
        {
            const unsigned int xb = f2bf_rne(xv);
            xs.u = xb | (xb << 16);
        }

        v8f acc0 = {}, acc1 = {}, acc2 = {}, acc3 = {};

        // Double-buffered A fragment: next k-step writes the other buffer, so the
        // WMMA->VALU WAR hazard (4 co-exec slots for bf16 WMMA) is covered by the
        // intervening pk_fma/pk_max stream instead of v_nops.
        Frag a[2];
#pragma unroll
        for (int s = 0; s < 4; ++s) {
            Frag& af = a[s & 1];
#pragma unroll
            for (int v = 0; v < 8; ++v) {
                Pk2 h;
                h.v = w1r[s * 8 + v].v * xs.v + b1r[s * 8 + v].v; // v_pk_fma_bf16
                af.u[v] = pk_relu_bf16(h.u);                      // v_pk_max_num_bf16
            }
            acc0 = __builtin_amdgcn_wmma_f32_16x16x32_bf16(false, af.v, false, bfr[s * 4 + 0].v,
                                                           (short)0, acc0, false, false);
            acc1 = __builtin_amdgcn_wmma_f32_16x16x32_bf16(false, af.v, false, bfr[s * 4 + 1].v,
                                                           (short)0, acc1, false, false);
            acc2 = __builtin_amdgcn_wmma_f32_16x16x32_bf16(false, af.v, false, bfr[s * 4 + 2].v,
                                                           (short)0, acc2, false, false);
            acc3 = __builtin_amdgcn_wmma_f32_16x16x32_bf16(false, af.v, false, bfr[s * 4 + 3].v,
                                                           (short)0, acc3, false, false);
        }

        // ---- Epilogue: +b2, masked embedding replacement, NT streaming stores ----
        // D layout: VGPR g -> row = base + g + 8*grp, col = 16n + lr
        int mrow[8];
#pragma unroll
        for (int g = 0; g < 8; ++g)
            mrow[g] = sMask[wave * 16 + g + grp * 8];

#pragma unroll
        for (int n = 0; n < 4; ++n) {
            const int   col = n * 16 + lr;
            const float bb  = sB2[col];
            const float ev  = sEmb[col];
            const v8f   acc = (n == 0) ? acc0 : (n == 1) ? acc1 : (n == 2) ? acc2 : acc3;
#pragma unroll
            for (int g = 0; g < 8; ++g) {
                const size_t r2  = base + g + grp * 8;
                const float  val = (mrow[g] == 0) ? ev : (acc[g] + bb);
                __builtin_nontemporal_store(val, out + (r2 * FDIM + f) * DDIM + col);
            }
        }
    }
}

extern "C" void kernel_launch(void* const* d_in, const int* in_sizes, int n_in,
                              void* d_out, int out_size, void* d_ws, size_t ws_size,
                              hipStream_t stream) {
    const float* x      = (const float*)d_in[0];
    const int*   x_mask = (const int*)  d_in[1];
    const float* W1     = (const float*)d_in[2];
    const float* b1     = (const float*)d_in[3];
    const float* W2     = (const float*)d_in[4];
    const float* b2     = (const float*)d_in[5];
    const float* emb    = (const float*)d_in[6];
    float* out          = (float*)d_out;

    dim3 grid(FDIM, /*B chunks=*/16384 / ROWS_PER_WG); // (256, 8)
    emb_mlp_wmma_kernel<<<grid, WG_THREADS, 0, stream>>>(x, x_mask, W1, b1, W2, b2, emb, out);
}
